// Behavior_34059090657285
// MI455X (gfx1250) — matrix operand order repaired and checked by reading, not compiled, not run on hardware
//
#include <hip/hip_runtime.h>
#include <hip/hip_bf16.h>

typedef __attribute__((ext_vector_type(16))) __bf16 v16bf;
typedef __attribute__((ext_vector_type(8)))  float  v8f;
typedef __attribute__((ext_vector_type(4))) unsigned int u32x4;
typedef __attribute__((ext_vector_type(8))) int i32x8;
typedef __attribute__((ext_vector_type(4))) int i32x4;

#define H14 14
#define W14 14
#define B16 16
#define CIN 1024
#define CMID 256
#define NCLS 27
#define M_SPAT (B16 * H14 * W14)        // 3136
#define N_ROI  (B16 * 5 * H14 * W14)    // 15680
#define KFC    (CMID * 9)               // 2304

#if defined(__has_builtin)
#if __has_builtin(__builtin_amdgcn_tensor_load_to_lds)
#define HAVE_TDM 1
#endif
#if __has_builtin(__builtin_amdgcn_s_wait_tensorcnt)
#define HAVE_TENSORCNT 1
#endif
#endif

__device__ __forceinline__ unsigned short f32_to_bf16_rn(float f) {
    unsigned int u = __float_as_uint(f);
    unsigned int lsb = (u >> 16) & 1u;
    u += 0x7FFFu + lsb;
    return (unsigned short)(u >> 16);
}

#ifdef HAVE_TDM
// Issue a TDM load of a [rows x 32] bf16 tile (row stride = row_stride_elems)
// from global into LDS at byte offset lds_byte_off. Uniform args; wave-level op.
// amdgpu-toolchain (clang-23) 6-arg form: (g0, g1, g2, g3, g4, cpol).
__device__ __forceinline__ void tdm_load_tile_b32cols(
    unsigned lds_byte_off, const unsigned short* gptr, unsigned rows,
    unsigned row_stride_elems) {
    unsigned long long ga = (unsigned long long)(size_t)gptr;
    u32x4 g0;
    g0[0] = 1u;                                   // count=1, user mode
    g0[1] = lds_byte_off;                         // lds_addr (bytes)
    g0[2] = (unsigned)(ga & 0xFFFFFFFFull);       // global_addr[31:0]
    g0[3] = (unsigned)((ga >> 32) & 0x1FFFFFFull) // global_addr[56:32]
          | (2u << 30);                           // type=2 ("image")
    i32x8 g1;
    g1[0] = (int)(1u << 16);                      // data_size=1 (2 bytes); wg_mask=0
    g1[1] = (int)(32u << 16);                     // tensor_dim0[15:0]=32 (atomic_barrier_addr=0)
    g1[2] = (int)(rows << 16);                    // tensor_dim0[31:16]=0 | tensor_dim1[15:0]=rows
    g1[3] = (int)(32u << 16);                     // tensor_dim1 hi=0 | tile_dim0=32
    g1[4] = (int)(rows & 0xFFFFu);                // tile_dim1=rows, tile_dim2=0
    g1[5] = (int)row_stride_elems;                // tensor_dim0_stride[31:0]
    g1[6] = 0;                                    // stride hi, tensor_dim1_stride lo
    g1[7] = 0;
    i32x4 z4 = {0, 0, 0, 0};
    i32x8 z8 = {0, 0, 0, 0, 0, 0, 0, 0};
    __builtin_amdgcn_tensor_load_to_lds(g0, g1, z4, z4, z8, 0);
}
#endif

__device__ __forceinline__ void wait_tensorcnt0() {
#ifdef HAVE_TENSORCNT
    __builtin_amdgcn_s_wait_tensorcnt((short)0);
#else
    asm volatile("s_wait_tensorcnt 0x0" ::: "memory");
#endif
}

// ---------------- prep kernels ----------------

__global__ void bn_prep_kernel(const float* __restrict__ g, const float* __restrict__ b,
                               const float* __restrict__ m, const float* __restrict__ v,
                               float* __restrict__ scale, float* __restrict__ bias, int n) {
    int i = blockIdx.x * blockDim.x + threadIdx.x;
    if (i < n) {
        float s = g[i] * rsqrtf(v[i] + 1e-5f);
        scale[i] = s;
        bias[i]  = b[i] - m[i] * s;
    }
}

__global__ void f32_to_bf16_kernel(const float* __restrict__ src,
                                   unsigned short* __restrict__ dst, int n) {
    int i = blockIdx.x * blockDim.x + threadIdx.x;
    if (i < n) dst[i] = f32_to_bf16_rn(src[i]);
}

// NCHW f32 -> [M=3136, K=1024] bf16 row-major (NHWC flattened)
__global__ void convert_x_kernel(const float* __restrict__ fmap,
                                 unsigned short* __restrict__ Xb) {
    int idx = blockIdx.x * blockDim.x + threadIdx.x;
    if (idx >= M_SPAT * CIN) return;
    int c = idx & (CIN - 1);
    int rest = idx >> 10;           // b*196 + s
    int s = rest % (H14 * W14);
    int b = rest / (H14 * W14);
    Xb[idx] = f32_to_bf16_rn(fmap[((size_t)b * CIN + c) * (H14 * W14) + s]);
}

// wfc [27,2304] f32 -> [32,2304] bf16 (zero-padded rows 27..31)
__global__ void convert_wfc_kernel(const float* __restrict__ wfc,
                                   unsigned short* __restrict__ dst) {
    int idx = blockIdx.x * blockDim.x + threadIdx.x;
    if (idx >= 32 * KFC) return;
    int r = idx / KFC;
    float val = (r < NCLS) ? wfc[idx] : 0.0f;
    dst[idx] = f32_to_bf16_rn(val);
}

// ---------------- WMMA GEMM + fused BN epilogue ----------------
// C[M,N] = A[M,K](bf16) * B[N,K](bf16, row n holds K)  then y = c*scale[n]+bias[n]
// Block: 128(M) x 64(N), 8 waves, each wave: 16(M) x 64(N) via 4 frags, K step 32.
// A staged via VGPRs (rows may overrun M into padded slack; epilogue masks).
// B tile (weights, always in-bounds) staged via the Tensor Data Mover.
// Dynamic LDS layout: sA @ 0 (8192 B), sB @ 8192 (4096 B).

template <bool OUT_BF16>
__global__ __launch_bounds__(256) void gemm_bn_kernel(
    const unsigned short* __restrict__ A, const unsigned short* __restrict__ Bw,
    const float* __restrict__ scale, const float* __restrict__ bias,
    void* __restrict__ Cout, int M, int N, int K) {
    extern __shared__ unsigned short smem[];
    unsigned short* sA = smem;             // 128*32
    unsigned short* sB = smem + 128 * 32;  // 64*32 @ byte offset 8192

    const int tid = threadIdx.x;
    const int wv  = tid >> 5;
    const int ln  = tid & 31;
    const int m0  = blockIdx.x * 128;
    const int n0  = blockIdx.y * 64;
    const int kh  = ln >> 4;
    const int lr  = ln & 15;

    v8f acc[4] = {};

    const int ar  = tid >> 1;                 // A row handled by this thread (pair of uint4)
    const int ac  = (tid & 1) << 4;           // 0 or 16 (elements)
    const unsigned short* Arow = A + (size_t)(m0 + ar) * K + ac;

    for (int k0 = 0; k0 < K; k0 += 32) {
        __syncthreads();
#ifdef HAVE_TDM
        if (tid < 32) {
            tdm_load_tile_b32cols(8192u, Bw + (size_t)n0 * K + k0, 64u, (unsigned)K);
        }
#else
        {
            int r   = tid >> 2;
            int col = (tid & 3) << 3;
            *(uint4*)(sB + r * 32 + col) =
                *(const uint4*)(Bw + (size_t)(n0 + r) * K + k0 + col);
        }
#endif
        // stage A: 128x32 bf16; each thread copies 16 contiguous bf16
        {
            const unsigned short* p = Arow + k0;
            uint4 v0 = *(const uint4*)p;
            uint4 v1 = *(const uint4*)(p + 8);
            __builtin_prefetch(p + 64, 0, 1);
            *(uint4*)(sA + ar * 32 + ac)     = v0;
            *(uint4*)(sA + ar * 32 + ac + 8) = v1;
        }
#ifdef HAVE_TDM
        if (tid < 32) wait_tensorcnt0();
#endif
        __syncthreads();

        // A fragment: lane lr = row, elems 0..7 -> K=8*kh.., elems 8..15 -> K=16+8*kh..
        union { v16bf f; uint4 u[2]; } fa;
        {
            const unsigned short* base = sA + ((wv << 4) + lr) * 32;
            fa.u[0] = *(const uint4*)(base + 8 * kh);
            fa.u[1] = *(const uint4*)(base + 16 + 8 * kh);
        }
#pragma unroll
        for (int nf = 0; nf < 4; ++nf) {
            // B fragment: lane lr = col n, elems 0..7 -> K=8*kh.., elems 8..15 -> K=16+8*kh..
            union { v16bf f; uint4 u[2]; } fb;
            const unsigned short* base = sB + ((nf << 4) + lr) * 32 + 8 * kh;
            fb.u[0] = *(const uint4*)(base);
            fb.u[1] = *(const uint4*)(base + 16);
            acc[nf] = __builtin_amdgcn_wmma_f32_16x16x32_bf16(
                false, fa.f, false, fb.f, (short)0, acc[nf], false, false);
        }
    }

    // epilogue: C/D layout -> lane ln, vgpr v: m = v + 8*(ln>>4), n = ln&15
#pragma unroll
    for (int nf = 0; nf < 4; ++nf) {
        int n = n0 + (nf << 4) + lr;
        float sc = scale[n], bi = bias[n];
#pragma unroll
        for (int v = 0; v < 8; ++v) {
            int m = m0 + (wv << 4) + v + (kh << 3);
            if (m < M) {
                float y = acc[nf][v] * sc + bi;
                if (OUT_BF16)
                    ((unsigned short*)Cout)[(size_t)m * N + n] = f32_to_bf16_rn(y);
                else
                    ((float*)Cout)[(size_t)m * N + n] = y;
            }
        }
    }
}

// ---------------- ROI-align (torchvision aligned=False, GRID=2, OUT=3) ----------------

__global__ __launch_bounds__(256) void roi_align_kernel(
    const float* __restrict__ boxes, const float* __restrict__ f256,
    unsigned short* __restrict__ feats) {
    __shared__ float sw[36][4];
    __shared__ int   si[36][4];

    const int R = blockIdx.x;               // 0..15679
    const int b = R / (5 * H14 * W14);
    const int t = threadIdx.x;

    if (t < 36) {
        const float* bx = boxes + (size_t)R * 4;
        float x1 = bx[0], y1 = bx[1], x2 = bx[2], y2 = bx[3];
        float rw = fmaxf(x2 - x1, 1.0f);
        float rh = fmaxf(y2 - y1, 1.0f);
        int p = t / 6, q = t % 6;
        float offy = (float)(p >> 1) + ((float)(p & 1) + 0.5f) * 0.5f;
        float offx = (float)(q >> 1) + ((float)(q & 1) + 0.5f) * 0.5f;
        float y = y1 + offy * (rh * (1.0f / 3.0f));
        float x = x1 + offx * (rw * (1.0f / 3.0f));
        bool valid = (y >= -1.0f) && (y <= 14.0f) && (x >= -1.0f) && (x <= 14.0f);
        y = fmaxf(y, 0.0f);
        x = fmaxf(x, 0.0f);
        int yl = min((int)floorf(y), 13);
        int xl = min((int)floorf(x), 13);
        int yh = min(yl + 1, 13);
        int xh = min(xl + 1, 13);
        if (yl == 13) y = 13.0f;
        if (xl == 13) x = 13.0f;
        float ly = y - (float)yl, lx = x - (float)xl;
        float hy = 1.0f - ly, hx = 1.0f - lx;
        float s = valid ? 1.0f : 0.0f;
        sw[t][0] = hy * hx * s;  sw[t][1] = hy * lx * s;
        sw[t][2] = ly * hx * s;  sw[t][3] = ly * lx * s;
        si[t][0] = yl * W14 + xl;  si[t][1] = yl * W14 + xh;
        si[t][2] = yh * W14 + xl;  si[t][3] = yh * W14 + xh;
    }
    __syncthreads();

    const float* fb = f256 + (size_t)b * (H14 * W14) * CMID;
    const int c = t;
    unsigned short* fr = feats + (size_t)R * KFC + c * 9;
#pragma unroll
    for (int by = 0; by < 3; ++by)
#pragma unroll
        for (int bx2 = 0; bx2 < 3; ++bx2) {
            float acc = 0.0f;
#pragma unroll
            for (int gy = 0; gy < 2; ++gy)
#pragma unroll
                for (int gx = 0; gx < 2; ++gx) {
                    int s = (by * 2 + gy) * 6 + (bx2 * 2 + gx);
                    acc += sw[s][0] * fb[si[s][0] * CMID + c]
                         + sw[s][1] * fb[si[s][1] * CMID + c]
                         + sw[s][2] * fb[si[s][2] * CMID + c]
                         + sw[s][3] * fb[si[s][3] * CMID + c];
                }
            fr[by * 3 + bx2] = f32_to_bf16_rn(acc * 0.25f);
        }
}

// ---------------- FC: [15680 x 2304] @ wfc^T (padded to 32 cols) + bias ----------------
// Dynamic LDS: sA @ 0 (8192 B), sB @ 8192 (2048 B).

__global__ __launch_bounds__(256) void fc_kernel(
    const unsigned short* __restrict__ A, const unsigned short* __restrict__ Bw,
    const float* __restrict__ bfc, float* __restrict__ out, int M, int K) {
    extern __shared__ unsigned short smem[];
    unsigned short* sA = smem;             // 128*32
    unsigned short* sB = smem + 128 * 32;  // 32*32 @ byte offset 8192

    const int tid = threadIdx.x;
    const int wv  = tid >> 5;
    const int ln  = tid & 31;
    const int m0  = blockIdx.x * 128;
    const int kh  = ln >> 4;
    const int lr  = ln & 15;

    v8f acc[2] = {};

    const int ar = tid >> 1;
    const int ac = (tid & 1) << 4;
    const unsigned short* Arow = A + (size_t)(m0 + ar) * K + ac;

    for (int k0 = 0; k0 < K; k0 += 32) {
        __syncthreads();
#ifdef HAVE_TDM
        if (tid < 32) {
            tdm_load_tile_b32cols(8192u, Bw + k0, 32u, (unsigned)K);
        }
#else
        if (tid < 128) {
            int r   = tid >> 2;
            int col = (tid & 3) << 3;
            *(uint4*)(sB + r * 32 + col) =
                *(const uint4*)(Bw + (size_t)r * K + k0 + col);
        }
#endif
        {
            const unsigned short* p = Arow + k0;
            uint4 v0 = *(const uint4*)p;
            uint4 v1 = *(const uint4*)(p + 8);
            __builtin_prefetch(p + 64, 0, 1);
            *(uint4*)(sA + ar * 32 + ac)     = v0;
            *(uint4*)(sA + ar * 32 + ac + 8) = v1;
        }
#ifdef HAVE_TDM
        if (tid < 32) wait_tensorcnt0();
#endif
        __syncthreads();

        union { v16bf f; uint4 u[2]; } fa;
        {
            const unsigned short* base = sA + ((wv << 4) + lr) * 32;
            fa.u[0] = *(const uint4*)(base + 8 * kh);
            fa.u[1] = *(const uint4*)(base + 16 + 8 * kh);
        }
#pragma unroll
        for (int nf = 0; nf < 2; ++nf) {
            union { v16bf f; uint4 u[2]; } fb;
            const unsigned short* base = sB + ((nf << 4) + lr) * 32 + 8 * kh;
            fb.u[0] = *(const uint4*)(base);
            fb.u[1] = *(const uint4*)(base + 16);
            acc[nf] = __builtin_amdgcn_wmma_f32_16x16x32_bf16(
                false, fa.f, false, fb.f, (short)0, acc[nf], false, false);
        }
    }

#pragma unroll
    for (int nf = 0; nf < 2; ++nf) {
        int n = (nf << 4) + lr;
        if (n >= NCLS) continue;
        float bi = bfc[n];
#pragma unroll
        for (int v = 0; v < 8; ++v) {
            int m = m0 + (wv << 4) + v + (kh << 3);
            if (m < M) out[(size_t)m * NCLS + n] = acc[nf][v] + bi;
        }
    }
}

// ---------------- launcher ----------------

extern "C" void kernel_launch(void* const* d_in, const int* in_sizes, int n_in,
                              void* d_out, int out_size, void* d_ws, size_t ws_size,
                              hipStream_t stream) {
    const float* boxes = (const float*)d_in[0];
    const float* fmap  = (const float*)d_in[1];
    const float* w1    = (const float*)d_in[2];
    const float* g1    = (const float*)d_in[3];
    const float* b1    = (const float*)d_in[4];
    const float* m1    = (const float*)d_in[5];
    const float* v1    = (const float*)d_in[6];
    const float* w2    = (const float*)d_in[7];
    const float* g2    = (const float*)d_in[8];
    const float* b2    = (const float*)d_in[9];
    const float* m2    = (const float*)d_in[10];
    const float* v2    = (const float*)d_in[11];
    const float* wfc   = (const float*)d_in[12];
    const float* bfc   = (const float*)d_in[13];
    float* out = (float*)d_out;

    // carve workspace (256-byte aligned slabs); A-operand slabs get 128 rows of
    // slack so the GEMM K-loop can load the last (partial) M-tile unguarded.
    char* ws = (char*)d_ws;
    size_t off = 0;
    auto carve = [&](size_t bytes) -> char* {
        char* p = ws + off;
        off += (bytes + 255) & ~(size_t)255;
        return p;
    };
    float* scale1 = (float*)carve(CIN * 4);
    float* bias1  = (float*)carve(CIN * 4);
    float* scale2 = (float*)carve(CMID * 4);
    float* bias2  = (float*)carve(CMID * 4);
    unsigned short* Xb   = (unsigned short*)carve((size_t)(M_SPAT + 128) * CIN * 2);
    unsigned short* W1b  = (unsigned short*)carve((size_t)CIN * CIN * 2);
    unsigned short* W2b  = (unsigned short*)carve((size_t)CMID * CIN * 2);
    unsigned short* Wfcb = (unsigned short*)carve((size_t)32 * KFC * 2);
    unsigned short* H1b  = (unsigned short*)carve((size_t)(M_SPAT + 128) * CIN * 2);
    float*          F256 = (float*)carve((size_t)M_SPAT * CMID * 4);
    unsigned short* Ft   = (unsigned short*)carve((size_t)(N_ROI + 128) * KFC * 2);
    (void)ws_size; (void)n_in; (void)in_sizes; (void)out_size;

    // BN fold
    bn_prep_kernel<<<(CIN + 255) / 256, 256, 0, stream>>>(g1, b1, m1, v1, scale1, bias1, CIN);
    bn_prep_kernel<<<(CMID + 255) / 256, 256, 0, stream>>>(g2, b2, m2, v2, scale2, bias2, CMID);

    // precision conversion
    f32_to_bf16_kernel<<<(CIN * CIN + 255) / 256, 256, 0, stream>>>(w1, W1b, CIN * CIN);
    f32_to_bf16_kernel<<<(CMID * CIN + 255) / 256, 256, 0, stream>>>(w2, W2b, CMID * CIN);
    convert_x_kernel<<<(M_SPAT * CIN + 255) / 256, 256, 0, stream>>>(fmap, Xb);
    convert_wfc_kernel<<<(32 * KFC + 255) / 256, 256, 0, stream>>>(wfc, Wfcb);

    const size_t lds_gemm = (size_t)(128 * 32 + 64 * 32) * 2;  // 12288 B
    const size_t lds_fc   = (size_t)(128 * 32 + 32 * 32) * 2;  // 10240 B

    // GEMM1: [3136,1024] x w1^T -> bf16 H1 (BN fused)
    gemm_bn_kernel<true><<<dim3((M_SPAT + 127) / 128, CIN / 64), 256, lds_gemm, stream>>>(
        Xb, W1b, scale1, bias1, H1b, M_SPAT, CIN, CIN);
    // GEMM2: H1 x w2^T -> f32 [3136,256] (BN fused)
    gemm_bn_kernel<false><<<dim3((M_SPAT + 127) / 128, CMID / 64), 256, lds_gemm, stream>>>(
        H1b, W2b, scale2, bias2, F256, M_SPAT, CMID, CIN);

    // ROI-align -> bf16 feats [15680, 2304]
    roi_align_kernel<<<N_ROI, 256, 0, stream>>>(boxes, F256, Ft);

    // FC -> f32 logits [15680, 27] == final output memory layout
    fc_kernel<<<(N_ROI + 127) / 128, 256, lds_fc, stream>>>(Ft, Wfcb, bfc, out, N_ROI, KFC);
}
